// Router_5222680232633
// MI455X (gfx1250) — matrix operand order repaired and checked by reading, not compiled, not run on hardware
//
#include <hip/hip_runtime.h>
#include <hip/hip_bf16.h>

#define R_ 8
#define N_ 8192
#define D_ 2048
#define A_ 128
#define LN_EPS 1e-5f

typedef __attribute__((ext_vector_type(16))) __bf16 v16bf;
typedef __attribute__((ext_vector_type(8)))  float  v8f;

__device__ __forceinline__ unsigned short f2bf(float f) {
    unsigned u = __float_as_uint(f);
    unsigned r = u + 0x7FFFu + ((u >> 16) & 1u);   // round-to-nearest-even
    return (unsigned short)(r >> 16);
}
__device__ __forceinline__ unsigned pack2(float lo, float hi) {
    return (unsigned)f2bf(lo) | ((unsigned)f2bf(hi) << 16);
}

// ---------------- Kernel 1: per-row mean / rstd of x ----------------
__global__ __launch_bounds__(256) void row_stats_kernel(
    const float* __restrict__ x, float* __restrict__ mean, float* __restrict__ rstd) {
    int n = blockIdx.x, t = threadIdx.x;
    const float4* row = (const float4*)(x + (size_t)n * D_);
    float s = 0.f, q = 0.f;
    #pragma unroll
    for (int i = t; i < D_ / 4; i += 256) {
        float4 v = row[i];
        s += v.x + v.y + v.z + v.w;
        q += v.x * v.x + v.y * v.y + v.z * v.z + v.w * v.w;
    }
    #pragma unroll
    for (int m = 16; m >= 1; m >>= 1) { s += __shfl_xor(s, m, 32); q += __shfl_xor(q, m, 32); }
    __shared__ float ss[8], qq[8];
    if ((t & 31) == 0) { ss[t >> 5] = s; qq[t >> 5] = q; }
    __syncthreads();
    if (t == 0) {
        float S = 0.f, Q = 0.f;
        #pragma unroll
        for (int i = 0; i < 8; ++i) { S += ss[i]; Q += qq[i]; }
        float m = S / (float)D_;
        float v = Q / (float)D_ - m * m;
        mean[n] = m;
        rstd[n] = rsqrtf(v + LN_EPS);
    }
}

// ---------------- Kernel 2: per-(router, column) stats of W over D ----------------
__global__ __launch_bounds__(256) void col_stats_kernel(
    const float* __restrict__ W, float* __restrict__ mu, float* __restrict__ rs) {
    int r = blockIdx.x, t = threadIdx.x;
    int a = t & (A_ - 1), dh = t >> 7;               // 2 partial sums per column
    const float* Wr = W + (size_t)r * D_ * A_;
    float s = 0.f, q = 0.f;
    for (int d = dh * (D_ / 2); d < (dh + 1) * (D_ / 2); ++d) {
        float w = Wr[(size_t)d * A_ + a];
        s += w; q += w * w;
    }
    __shared__ float ss[256], qq[256];
    ss[t] = s; qq[t] = q;
    __syncthreads();
    if (dh == 0) {
        float S = ss[t] + ss[t + 128], Q = qq[t] + qq[t + 128];
        float m = S / (float)D_;
        float v = Q / (float)D_ - m * m;
        mu[r * A_ + a] = m;
        rs[r * A_ + a] = rsqrtf(v + LN_EPS);
    }
}

// ---------------- Kernel 3: fused LN + bf16 WMMA GEMM + softmax ----------------
// Block: 256 threads (8 waves). Tile: 128 rows x 128 cols, K-chunks of 32.
__global__ __launch_bounds__(256) void router_main_kernel(
    const float* __restrict__ x,  const float* __restrict__ ln_w, const float* __restrict__ ln_b,
    const float* __restrict__ W,  const float* __restrict__ g,    const float* __restrict__ beta,
    const float* __restrict__ bias,
    const float* __restrict__ rmean, const float* __restrict__ rrstd,
    const float* __restrict__ cmu,   const float* __restrict__ crs,
    float* __restrict__ newx, float* __restrict__ logits, float* __restrict__ probs)
{
    const int r  = blockIdx.y;
    const int n0 = blockIdx.x * 128;
    const int t  = threadIdx.x;

    // A-writer mapping: thread -> (row, 16-wide k-half)
    const int arow = t >> 1;
    const int ah   = t & 1;
    // B-writer mapping: thread -> (even k-row pair, 8 columns)
    const int bkk  = (t >> 4) * 2;        // 0..30 even
    const int ba0  = (t & 15) * 8;        // 0..120

    __shared__ __align__(32) unsigned short As[128 * 32];   // y tile, A-frag swizzled
    __shared__ __align__(32) unsigned short Bs[32 * 128];   // W tile, B-frag swizzled
    __shared__ float tsum[128], usum[128];

    const float rm = rmean[n0 + arow];
    const float rs = rrstd[n0 + arow];
    const float* xrow  = x + (size_t)(n0 + arow) * D_;
    float*       nxrow = newx + ((size_t)r * N_ + (n0 + arow)) * D_;
    const float* lw = ln_w + (size_t)r * D_;
    const float* lb = ln_b + (size_t)r * D_;
    const float* gg = g    + (size_t)r * D_;
    const float* bb = beta + (size_t)r * D_;
    const float* Wr = W + (size_t)r * D_ * A_;

    float ty = 0.f, tu = 0.f;   // per-(row,half) partials of sum(y), sum(nx*beta)

    const int lane = t & 31, w = t >> 5;
    const int lrow = lane & 15, hi = lane >> 4;

    v8f acc[8];
    #pragma unroll
    for (int i = 0; i < 8; ++i) acc[i] = v8f{0.f,0.f,0.f,0.f,0.f,0.f,0.f,0.f};

    const unsigned short* aptr = &As[(((w * 16 + lrow) * 2) + hi) * 16];

    for (int k0 = 0; k0 < D_; k0 += 32) {
        // ---- stage A tile: new_x (write out, f32) and y=new_x*g (-> LDS, bf16, swizzled)
        {
            const int kb = k0 + ah * 16;
            #pragma unroll
            for (int j4 = 0; j4 < 4; ++j4) {
                float4 xv = *(const float4*)(xrow + kb + j4 * 4);
                float4 wv = *(const float4*)(lw   + kb + j4 * 4);
                float4 bv = *(const float4*)(lb   + kb + j4 * 4);
                float4 gv = *(const float4*)(gg   + kb + j4 * 4);
                float4 ev = *(const float4*)(bb   + kb + j4 * 4);
                float nx0 = (xv.x - rm) * rs * wv.x + bv.x;
                float nx1 = (xv.y - rm) * rs * wv.y + bv.y;
                float nx2 = (xv.z - rm) * rs * wv.z + bv.z;
                float nx3 = (xv.w - rm) * rs * wv.w + bv.w;
                *(float4*)(nxrow + kb + j4 * 4) = make_float4(nx0, nx1, nx2, nx3);
                float y0 = nx0 * gv.x, y1 = nx1 * gv.y, y2 = nx2 * gv.z, y3 = nx3 * gv.w;
                ty += y0 + y1 + y2 + y3;
                tu += nx0 * ev.x + nx1 * ev.y + nx2 * ev.z + nx3 * ev.w;
                // swizzle into A-fragment layout (16-bit A 16x32, wave32)
                int kk0 = ah * 16 + j4 * 4;                      // even, multiple of 4
                int h   = (kk0 >> 3) & 1;
                int s   = (kk0 & 7) + 8 * (kk0 >> 4);            // even
                unsigned* dst = (unsigned*)As + ((arow * 2 + h) * 8 + (s >> 1));
                dst[0] = pack2(y0, y1);
                dst[1] = pack2(y2, y3);
            }
        }
        // ---- stage B tile: raw W -> bf16 -> LDS, swizzled B-fragment layout
        {
            const float* w0 = Wr + (size_t)(k0 + bkk) * A_ + ba0;
            const float* w1 = w0 + A_;
            const int slot = (bkk & 7) + 8 * (bkk >> 4);        // even
            const int loff = (bkk & 8) ? 16 : 0;
            #pragma unroll
            for (int j4 = 0; j4 < 2; ++j4) {
                float4 p = *(const float4*)(w0 + j4 * 4);
                float4 q = *(const float4*)(w1 + j4 * 4);
                float pe[4] = {p.x, p.y, p.z, p.w};
                float qe[4] = {q.x, q.y, q.z, q.w};
                #pragma unroll
                for (int c = 0; c < 4; ++c) {
                    int a    = ba0 + j4 * 4 + c;
                    int tile = a >> 4;
                    int lb_  = (a & 15) + loff;
                    ((unsigned*)Bs)[(tile * 32 + lb_) * 8 + (slot >> 1)] = pack2(pe[c], qe[c]);
                }
            }
        }
        __syncthreads();
        // ---- WMMA: each wave does 16x128 x 32-K via 8 x v_wmma_f32_16x16x32_bf16
        {
            v16bf af = *(const v16bf*)aptr;
            #pragma unroll
            for (int tb = 0; tb < 8; ++tb) {
                v16bf bf_ = *(const v16bf*)&Bs[(tb * 32 + lane) * 16];
                acc[tb] = __builtin_amdgcn_wmma_f32_16x16x32_bf16(
                    false, af, false, bf_, (short)0, acc[tb], false, false);
            }
        }
        __syncthreads();
    }

    // ---- row sums: combine thread pairs, publish to LDS
    float tyF = ty + __shfl_xor(ty, 1, 32);
    float tuF = tu + __shfl_xor(tu, 1, 32);
    if (ah == 0) { tsum[arow] = tyF; usum[arow] = tuF; }
    __syncthreads();

    // ---- epilogue: fold weight-LN scale, add bias, write logits, softmax, write probs
    const int rowbase = w * 16 + 8 * hi;     // C/D layout: lanes 0-15 rows 0-7, lanes 16-31 rows 8-15
    float trow[8], urow[8];
    #pragma unroll
    for (int j = 0; j < 8; ++j) { trow[j] = tsum[rowbase + j]; urow[j] = usum[rowbase + j]; }

    #pragma unroll
    for (int tb = 0; tb < 8; ++tb) {
        int a = tb * 16 + lrow;
        float mu  = cmu[r * A_ + a];
        float rsd = crs[r * A_ + a];
        float bs  = bias[r * A_ + a];
        #pragma unroll
        for (int j = 0; j < 8; ++j) {
            float lg = rsd * (acc[tb][j] - mu * trow[j]) + urow[j] + bs;
            acc[tb][j] = lg;
            size_t idx = ((size_t)r * N_ + (n0 + rowbase + j)) * A_ + a;
            logits[idx] = lg;
        }
    }
    #pragma unroll
    for (int j = 0; j < 8; ++j) {
        float mx = acc[0][j];
        #pragma unroll
        for (int tb = 1; tb < 8; ++tb) mx = fmaxf(mx, acc[tb][j]);
        #pragma unroll
        for (int m = 1; m < 16; m <<= 1) mx = fmaxf(mx, __shfl_xor(mx, m, 32));
        float s = 0.f;
        #pragma unroll
        for (int tb = 0; tb < 8; ++tb) {
            float e = expf(acc[tb][j] - mx);
            acc[tb][j] = e;
            s += e;
        }
        #pragma unroll
        for (int m = 1; m < 16; m <<= 1) s += __shfl_xor(s, m, 32);
        float inv = 1.0f / s;
        #pragma unroll
        for (int tb = 0; tb < 8; ++tb) {
            int a = tb * 16 + lrow;
            size_t idx = ((size_t)r * N_ + (n0 + rowbase + j)) * A_ + a;
            probs[idx] = acc[tb][j] * inv;
        }
    }
}

extern "C" void kernel_launch(void* const* d_in, const int* in_sizes, int n_in,
                              void* d_out, int out_size, void* d_ws, size_t ws_size,
                              hipStream_t stream) {
    const float* x      = (const float*)d_in[0];
    const float* ln_w   = (const float*)d_in[1];
    const float* ln_b   = (const float*)d_in[2];
    const float* W      = (const float*)d_in[3];
    const float* rw_ln_w= (const float*)d_in[4];
    const float* rw_ln_b= (const float*)d_in[5];
    const float* bias   = (const float*)d_in[6];

    float* out    = (float*)d_out;
    float* newx   = out;                                     // [8, 8192, 2048]
    float* logits = out + (size_t)R_ * N_ * D_;              // [8, 8192, 128]
    float* probs  = logits + (size_t)R_ * N_ * A_;           // [8, 8192, 128]

    float* ws    = (float*)d_ws;
    float* rmean = ws;                  // 8192
    float* rrstd = ws + N_;             // 8192
    float* cmu   = ws + 2 * N_;         // 1024
    float* crs   = ws + 2 * N_ + R_ * A_;

    row_stats_kernel<<<N_, 256, 0, stream>>>(x, rmean, rrstd);
    col_stats_kernel<<<R_, 256, 0, stream>>>(W, cmu, crs);

    dim3 grid(N_ / 128, R_);
    router_main_kernel<<<grid, 256, 0, stream>>>(
        x, ln_w, ln_b, W, rw_ln_w, rw_ln_b, bias,
        rmean, rrstd, cmu, crs, newx, logits, probs);
}
